// DyGODE_26001732010004
// MI455X (gfx1250) — matrix operand order repaired and checked by reading, not compiled, not run on hardware
//
#include <hip/hip_runtime.h>
#include <math.h>

typedef __attribute__((ext_vector_type(16))) _Float16 v16h;
typedef __attribute__((ext_vector_type(8)))  float    v8f;

namespace {
constexpr int B = 256, N = 207, L = 12, C = 32, RF = 19, ND = 40;
constexpr int MT2 = 14;         // M tiles padded to even (13 real + 1 zero)
constexpr int MBLK = 7;         // M tiles per wave (2 groups)
constexpr int KT = 7;           // ceil(207/32) K tiles
constexpr int NPAD = KT * 32;   // 224
constexpr float H1 = 0.4f;
constexpr float EPS = 1e-5f;

constexpr size_t AUP(size_t x) { return (x + 63) & ~size_t(63); }
// workspace offsets (in floats)
constexpr size_t O_N1   = 0;
constexpr size_t O_N2   = O_N1  + AUP((size_t)N * ND);
constexpr size_t O_ADJ  = O_N2  + AUP((size_t)N * ND);
constexpr size_t O_MASK = O_ADJ + AUP((size_t)N * N);
constexpr size_t O_ADJM = O_MASK+ AUP((size_t)N * N);
constexpr size_t O_MA   = O_ADJM+ AUP((size_t)N * N);
constexpr size_t O_MB   = O_MA  + AUP((size_t)N * N);
constexpr size_t O_M2A  = O_MB  + AUP((size_t)N * N);
constexpr size_t O_M2B  = O_M2A + AUP((size_t)N * N);
constexpr size_t O_M4A  = O_M2B + AUP((size_t)N * N);
constexpr size_t O_M4B  = O_M4A + AUP((size_t)N * N);
constexpr size_t O_STAT = O_M4B + AUP((size_t)N * N);                 // mean[B], rstd[B]
constexpr size_t O_AFRG = O_STAT+ AUP(2 * (size_t)B);                 // f16: 2*MT2*KT*512 halves
constexpr size_t O_H    = O_AFRG+ AUP((size_t)2 * MT2 * KT * 512 / 2);
constexpr size_t O_XG   = O_H   + AUP((size_t)B * C * N * RF);        // f16: B*26*KT*512 halves max
constexpr size_t O_Y1   = O_XG  + AUP((size_t)B * 26 * KT * 512 / 2);
constexpr size_t O_Y2   = O_Y1  + AUP((size_t)B * C * N * 13);
}

struct IncParams {
  const float* w1[4]; const float* b1[4];
  const float* w2[4]; const float* b2[4];
};

// ---------------- graph build ----------------
__global__ void build_nodes_k(const float* __restrict__ emb, const float* __restrict__ lw,
                              const float* __restrict__ lb, float* __restrict__ out) {
  int idx = blockIdx.x * blockDim.x + threadIdx.x;
  if (idx >= N * ND) return;
  int v = idx / ND, d = idx % ND;
  float acc = lb[d];
  for (int e = 0; e < ND; ++e) acc += emb[v * ND + e] * lw[d * ND + e];
  out[idx] = tanhf(3.0f * acc);
}

__global__ void adj_k(const float* __restrict__ n1, const float* __restrict__ n2,
                      float* __restrict__ adj) {
  int idx = blockIdx.x * blockDim.x + threadIdx.x;
  if (idx >= N * N) return;
  int v = idx / N, w = idx % N;
  float s = 0.f;
  for (int d = 0; d < ND; ++d)
    s += n1[v * ND + d] * n2[w * ND + d] - n2[v * ND + d] * n1[w * ND + d];
  float t = tanhf(3.0f * s);
  adj[idx] = t > 0.f ? t : 0.f;
}

// top-20 per row (jax tie-break: lowest index first)
__global__ void topk_k(const float* __restrict__ adj, float* __restrict__ mask,
                       float* __restrict__ adjm) {
  int row = blockIdx.x * blockDim.x + threadIdx.x;
  if (row >= N) return;
  float* mrow = mask + (size_t)row * N;
  const float* arow = adj + (size_t)row * N;
  for (int w = 0; w < N; ++w) mrow[w] = 0.f;
  for (int k = 0; k < 20; ++k) {
    int best = 0; float bv = -1e30f;
    for (int w = 0; w < N; ++w)
      if (mrow[w] == 0.f && arow[w] > bv) { bv = arow[w]; best = w; }
    mrow[best] = 1.f;
  }
  for (int w = 0; w < N; ++w) adjm[(size_t)row * N + w] = arow[w] * mrow[w];
}

// M = 0.875 I + 0.125 * rownorm(adjm + I), both directions
__global__ void buildM_k(const float* __restrict__ adjm, float* __restrict__ Ma,
                         float* __restrict__ Mb) {
  int idx = blockIdx.x * blockDim.x + threadIdx.x;
  if (idx >= N * N) return;
  int v = idx / N, w = idx % N;
  float rsA = 1.f, rsB = 1.f;
  for (int t = 0; t < N; ++t) { rsA += adjm[(size_t)v * N + t]; rsB += adjm[(size_t)t * N + v]; }
  float d = (v == w) ? 1.f : 0.f;
  Ma[idx] = 0.875f * d + 0.125f * (adjm[(size_t)v * N + w] + d) / rsA;
  Mb[idx] = 0.875f * d + 0.125f * (adjm[(size_t)w * N + v] + d) / rsB;
}

__global__ void matmulN_k(const float* __restrict__ A, const float* __restrict__ Bm,
                          float* __restrict__ Cm) {
  int idx = blockIdx.x * blockDim.x + threadIdx.x;
  if (idx >= N * N) return;
  int v = idx / N, w = idx % N;
  float acc = 0.f;
  for (int k = 0; k < N; ++k) acc += A[(size_t)v * N + k] * Bm[(size_t)k * N + w];
  Cm[idx] = acc;
}

// pack M4a/M4b into WMMA A-fragment order (16x16x32 f16 layout), padded to 14 row tiles
__global__ void packA_k(const float* __restrict__ M4a, const float* __restrict__ M4b,
                        _Float16* __restrict__ Afrag) {
  int idx = blockIdx.x * blockDim.x + threadIdx.x;
  if (idx >= 2 * MT2 * KT * 512) return;
  int e = idx & 15, lane = (idx >> 4) & 31;
  int q = idx >> 9;
  int kt = q % KT; q /= KT;
  int mt = q % MT2; int dir = q / MT2;
  int row = mt * 16 + (lane & 15);
  int b8 = (lane >> 4) * 8;
  int kk = (e < 8) ? (b8 + e) : (16 + b8 + (e - 8));
  int col = kt * 32 + kk;
  const float* M = dir ? M4b : M4a;
  float val = (row < N && col < N) ? M[(size_t)row * N + col] : 0.0f;
  Afrag[idx] = (_Float16)val;
}

// ---------------- temporal stack ----------------
__global__ void start_k(const float* __restrict__ x, const float* __restrict__ sw,
                        const float* __restrict__ sb, float* __restrict__ h) {
  int idx = blockIdx.x * blockDim.x + threadIdx.x;
  if (idx >= B * C * N * RF) return;
  int t = idx % RF;
  int v = (idx / RF) % N;
  int c = (idx / (RF * N)) % C;
  int b = idx / (RF * N * C);
  float acc = sb[c];
  int ts = t - (RF - L);
  if (ts >= 0) {
    acc += sw[c * 2 + 0] * x[(((size_t)b * 2 + 0) * N + v) * L + ts]
         + sw[c * 2 + 1] * x[(((size_t)b * 2 + 1) * N + v) * L + ts];
  }
  h[idx] = acc;
}

// inception (k=2,3,6,7, 8 ch each) + tanh*sigmoid gate, emit f16 in B-fragment order
__global__ void inception_k(const float* __restrict__ h, IncParams P,
                            _Float16* __restrict__ XG, int isl, int Lm, int JT) {
  int idx = blockIdx.x * blockDim.x + threadIdx.x;
  if (idx >= B * C * NPAD * Lm) return;
  int l = idx % Lm;
  int v = (idx / Lm) % NPAD;
  int c = (idx / (Lm * NPAD)) % C;
  int b = idx / (Lm * NPAD * C);
  float val = 0.f;
  if (v < N) {
    const int KS[4] = {2, 3, 6, 7};
    int g = c >> 3, co = c & 7, k = KS[g];
    float a1 = P.b1[g][co], a2 = P.b2[g][co];
    const float* hb = h + ((size_t)b * C * N + v) * RF + (RF - isl);
    int base = l + 7 - k;  // isl - Lm == 6 always
    for (int ci = 0; ci < C; ++ci) {
      const float* xr = hb + (size_t)ci * N * RF + base;
      const float* w1r = P.w1[g] + (size_t)(co * C + ci) * k;
      const float* w2r = P.w2[g] + (size_t)(co * C + ci) * k;
      for (int i = 0; i < k; ++i) {
        float xv = xr[i];
        a1 += w1r[i] * xv;
        a2 += w2r[i] * xv;
      }
    }
    val = tanhf(a1) * (1.0f / (1.0f + expf(-a2)));
  }
  int j = c * Lm + l;
  int jt = j >> 4, jc = j & 15;
  int kt = v >> 5, kk = v & 31;
  int lane = jc + (((kk >> 3) & 1) << 4);
  int elem = (kk & 7) + ((kk >> 4) << 3);
  XG[(((size_t)b * JT + jt) * KT + kt) * 512 + lane * 16 + elem] = (_Float16)val;
}

// WMMA node-propagation GEMM.
// blockIdx.y = dir*2 + mgroup: the block's 7 A row-tiles (49KB) are staged into LDS
// once by all 8 waves, then each wave owns one (b, jt) column strip:
//   per K-step: 1 global B-fragment load, 7 ds-fed WMMAs. 56 acc VGPRs -> no spills.
__global__ void __launch_bounds__(256) gemm_prop_k(
    const _Float16* __restrict__ Afrag, const _Float16* __restrict__ XG,
    float* __restrict__ Y1, float* __restrict__ Y2, int Lm, int JT) {
  __shared__ _Float16 As[MBLK * KT * 512];  // 50176 bytes
  int dir = blockIdx.y >> 1, mg = blockIdx.y & 1;

  {  // cooperative stage: contiguous 7-tile slab of Afrag -> LDS
    const uint4* src = (const uint4*)(Afrag + (size_t)(dir * MT2 + mg * MBLK) * KT * 512);
    uint4* dst = (uint4*)As;
    for (int i = threadIdx.x; i < MBLK * KT * 512 / 8; i += 256) dst[i] = src[i];
  }
  __syncthreads();

  int wid = blockIdx.x * 8 + (threadIdx.x >> 5);  // wave-uniform
  int lane = threadIdx.x & 31;
  if (wid >= B * JT) return;                      // whole-wave exit, EXEC stays all-1s
  int jt = wid % JT;
  int b = wid / JT;

  v8f acc[MBLK];
#pragma unroll
  for (int m = 0; m < MBLK; ++m) acc[m] = (v8f){};

  const _Float16* bbase = XG + (((size_t)b * JT + jt) * KT) * 512 + lane * 16;
  const _Float16* abase = As + lane * 16;

  for (int kt = 0; kt < KT; ++kt) {
    v16h bm = *(const v16h*)(bbase + (size_t)kt * 512);
    if (kt + 1 < KT) __builtin_prefetch(bbase + (size_t)(kt + 1) * 512, 0, 1);
#pragma unroll
    for (int m = 0; m < MBLK; ++m) {
      v16h a = *(const v16h*)(abase + (m * KT + kt) * 512);
      acc[m] = __builtin_amdgcn_wmma_f32_16x16x32_f16(false, a, false, bm, (short)0,
                                                      acc[m], false, false);
    }
  }

  int jc = lane & 15, half = lane >> 4;
  int j = jt * 16 + jc;
  int ch = j / Lm, l = j % Lm;
  float* Y = dir ? Y2 : Y1;
  size_t ybase = ((size_t)b * C + ch) * N;
#pragma unroll
  for (int m = 0; m < MBLK; ++m) {
    int vbase = (mg * MBLK + m) * 16 + half * 8;
#pragma unroll
    for (int r = 0; r < 8; ++r) {
      int v = vbase + r;
      if (v < N) Y[(ybase + v) * Lm + l] = acc[m][r];
    }
  }
}

// channel mix (g1,g2 conv1x1) + residual update h += H1 * f (front-padded)
__global__ void mix_update_k(const float* __restrict__ Y1, const float* __restrict__ Y2,
                             const float* __restrict__ g1w, const float* __restrict__ g1b,
                             const float* __restrict__ g2w, const float* __restrict__ g2b,
                             float* __restrict__ h, int Lm) {
  int idx = blockIdx.x * blockDim.x + threadIdx.x;
  if (idx >= B * C * N * Lm) return;
  int l = idx % Lm;
  int v = (idx / Lm) % N;
  int o = (idx / (Lm * N)) % C;
  int b = idx / (Lm * N * C);
  float acc = g1b[o] + g2b[o];
  for (int c = 0; c < C; ++c) {
    size_t yi = (((size_t)b * C + c) * N + v) * Lm + l;
    acc += g1w[o * C + c] * Y1[yi] + g2w[o * C + c] * Y2[yi];
  }
  h[(((size_t)b * C + o) * N + v) * RF + (RF - Lm) + l] += H1 * acc;
}

// ---------------- head ----------------
__global__ void stats_k(const float* __restrict__ h, float* __restrict__ stat) {
  __shared__ float s1[256], s2[256];
  int b = blockIdx.x, tid = threadIdx.x;
  float a = 0.f, q = 0.f;
  for (int i = tid; i < C * N; i += 256) {
    float x = h[((size_t)b * C * N + i) * RF + (RF - 1)];
    a += x; q += x * x;
  }
  s1[tid] = a; s2[tid] = q;
  __syncthreads();
  for (int s = 128; s > 0; s >>= 1) {
    if (tid < s) { s1[tid] += s1[tid + s]; s2[tid] += s2[tid + s]; }
    __syncthreads();
  }
  if (tid == 0) {
    float mean = s1[0] / (float)(C * N);
    float var = s2[0] / (float)(C * N) - mean * mean;
    stat[b] = mean;
    stat[B + b] = rsqrtf(var + EPS);
  }
}

__global__ void normalize_k(const float* __restrict__ h, const float* __restrict__ stat,
                            const float* __restrict__ affw, const float* __restrict__ affb,
                            float* __restrict__ nrm) {
  int idx = blockIdx.x * blockDim.x + threadIdx.x;
  if (idx >= B * C * N) return;
  int v = idx % N;
  int c = (idx / N) % C;
  int b = idx / (N * C);
  float x = h[(size_t)idx * RF + (RF - 1)];
  nrm[idx] = (x - stat[b]) * stat[B + b] * affw[c * N + v] + affb[c * N + v];
}

__global__ void fc_k(const float* __restrict__ in, const float* __restrict__ w,
                     const float* __restrict__ bias, float* __restrict__ out,
                     int O, int I, int do_relu) {
  int idx = blockIdx.x * blockDim.x + threadIdx.x;
  if (idx >= B * O * N) return;
  int v = idx % N;
  int o = (idx / N) % O;
  int b = idx / (N * O);
  float acc = bias[o];
  const float* ir = in + (size_t)b * I * N + v;
  const float* wr = w + (size_t)o * I;
  for (int c = 0; c < I; ++c) acc += wr[c] * ir[(size_t)c * N];
  if (do_relu) acc = fmaxf(acc, 0.f);
  out[idx] = acc;
}

#define LAUNCH(kern, n, ...) kern<<<((n) + 255) / 256, 256, 0, stream>>>(__VA_ARGS__)

extern "C" void kernel_launch(void* const* d_in, const int* in_sizes, int n_in,
                              void* d_out, int out_size, void* d_ws, size_t ws_size,
                              hipStream_t stream) {
  (void)in_sizes; (void)n_in; (void)out_size; (void)ws_size;
  const float* x       = (const float*)d_in[0];
  const float* emb1    = (const float*)d_in[1];
  const float* emb2    = (const float*)d_in[2];
  const float* lin1_w  = (const float*)d_in[3];
  const float* lin1_b  = (const float*)d_in[4];
  const float* lin2_w  = (const float*)d_in[5];
  const float* lin2_b  = (const float*)d_in[6];
  const float* start_w = (const float*)d_in[7];
  const float* start_b = (const float*)d_in[8];
  IncParams P;
  for (int g = 0; g < 4; ++g) {
    P.w1[g] = (const float*)d_in[9 + g];
    P.b1[g] = (const float*)d_in[13 + g];
    P.w2[g] = (const float*)d_in[17 + g];
    P.b2[g] = (const float*)d_in[21 + g];
  }
  const float* g1_w  = (const float*)d_in[25];
  const float* g1_b  = (const float*)d_in[26];
  const float* g2_w  = (const float*)d_in[27];
  const float* g2_b  = (const float*)d_in[28];
  const float* aff_w = (const float*)d_in[29];
  const float* aff_b = (const float*)d_in[30];
  const float* e0_w  = (const float*)d_in[31];
  const float* e0_b  = (const float*)d_in[32];
  const float* e1_w  = (const float*)d_in[33];
  const float* e1_b  = (const float*)d_in[34];
  const float* e2_w  = (const float*)d_in[35];
  const float* e2_b  = (const float*)d_in[36];

  float* ws = (float*)d_ws;
  float* n1   = ws + O_N1;
  float* n2   = ws + O_N2;
  float* adj  = ws + O_ADJ;
  float* mask = ws + O_MASK;
  float* adjm = ws + O_ADJM;
  float* Ma   = ws + O_MA;
  float* Mb   = ws + O_MB;
  float* M2a  = ws + O_M2A;
  float* M2b  = ws + O_M2B;
  float* M4a  = ws + O_M4A;
  float* M4b  = ws + O_M4B;
  float* stat = ws + O_STAT;
  _Float16* Afrag = (_Float16*)(ws + O_AFRG);
  float* h    = ws + O_H;
  _Float16* XG = (_Float16*)(ws + O_XG);
  float* Y1   = ws + O_Y1;
  float* Y2   = ws + O_Y2;
  // head temporaries overlay buffers that are dead by then
  float* nrm = Y1;                  // B*C*N
  float* r0  = Y2;                  // B*64*N
  float* r1  = (float*)(ws + O_XG); // B*128*N

  // graph build -> M^4 for both directions
  LAUNCH(build_nodes_k, N * ND, emb1, lin1_w, lin1_b, n1);
  LAUNCH(build_nodes_k, N * ND, emb2, lin2_w, lin2_b, n2);
  LAUNCH(adj_k, N * N, n1, n2, adj);
  LAUNCH(topk_k, N, adj, mask, adjm);
  LAUNCH(buildM_k, N * N, adjm, Ma, Mb);
  LAUNCH(matmulN_k, N * N, Ma, Ma, M2a);
  LAUNCH(matmulN_k, N * N, Mb, Mb, M2b);
  LAUNCH(matmulN_k, N * N, M2a, M2a, M4a);
  LAUNCH(matmulN_k, N * N, M2b, M2b, M4b);
  LAUNCH(packA_k, 2 * MT2 * KT * 512, M4a, M4b, Afrag);

  // start conv (with front zero-pad to RF)
  LAUNCH(start_k, B * C * N * RF, x, start_w, start_b, h);

  int isl = RF;
  for (int it = 0; it < 3; ++it) {
    int Lm = isl - 6;          // 13, 7, 1
    int JT = (C * Lm) / 16;    // 26, 14, 2
    LAUNCH(inception_k, B * C * NPAD * Lm, h, P, XG, isl, Lm, JT);
    dim3 ggrid((B * JT + 7) / 8, 4);   // y: dir*2 + mgroup
    gemm_prop_k<<<ggrid, 256, 0, stream>>>(Afrag, XG, Y1, Y2, Lm, JT);
    LAUNCH(mix_update_k, B * C * N * Lm, Y1, Y2, g1_w, g1_b, g2_w, g2_b, h, Lm);
    isl = Lm;
  }

  // head: layernorm + affine + 3x conv1x1
  stats_k<<<B, 256, 0, stream>>>(h, stat);
  LAUNCH(normalize_k, B * C * N, h, stat, aff_w, aff_b, nrm);
  LAUNCH(fc_k, B * 64 * N, nrm, e0_w, e0_b, r0, 64, 32, 1);
  LAUNCH(fc_k, B * 128 * N, r0, e1_w, e1_b, r1, 128, 64, 1);
  LAUNCH(fc_k, B * 12 * N, r1, e2_w, e2_b, (float*)d_out, 12, 128, 0);
}
